// REAM_25975962206424
// MI455X (gfx1250) — matrix operand order repaired and checked by reading, not compiled, not run on hardware
//
#include <hip/hip_runtime.h>
#include <stdint.h>

// Problem constants
#define BB 8
#define PP 4096
#define CC 768
#define NN 116
#define BP (BB * PP)          // 32768 rows
#define SEG (BB * NN)         // 928 segment rows

typedef __bf16 bf16;
typedef bf16  v16bf __attribute__((ext_vector_type(16)));
typedef float v8f   __attribute__((ext_vector_type(8)));

__device__ __forceinline__ unsigned short f2bf(float f) {
  unsigned u = __float_as_uint(f);
  u += 0x7FFFu + ((u >> 16) & 1u);
  return (unsigned short)(u >> 16);
}

__device__ __forceinline__ void atomicMaxF(float* addr, float v) {
  if (v >= 0.f) atomicMax((int*)addr, __float_as_int(v));
  else          atomicMin((unsigned int*)addr, (unsigned int)__float_as_int(v));
}

__device__ __forceinline__ void atomAddF(float* p, float v) {
  __hip_atomic_fetch_add(p, v, __ATOMIC_RELAXED, __HIP_MEMORY_SCOPE_AGENT);
}

__device__ __forceinline__ void waitAsync0() {
#if defined(__has_builtin) && __has_builtin(__builtin_amdgcn_s_wait_asynccnt)
  __builtin_amdgcn_s_wait_asynccnt(0);
#else
  asm volatile("s_wait_asynccnt 0" ::: "memory");
#endif
}

// async copy of 16 bytes global -> LDS (per lane), tracked by ASYNCcnt
__device__ __forceinline__ void asyncB128(unsigned ldsOff, const void* gptr) {
  asm volatile("global_load_async_to_lds_b128 %0, %1, off"
               :: "v"(ldsOff), "v"((unsigned long long)gptr) : "memory");
}
__device__ __forceinline__ void asyncB128_o16(unsigned ldsOff, const void* gptr) {
  asm volatile("global_load_async_to_lds_b128 %0, %1, off offset:16"
               :: "v"(ldsOff), "v"((unsigned long long)gptr) : "memory");
}

// ---------------------------------------------------------------------------
// init: m = -inf, denom = 0, xagg = 0
// ---------------------------------------------------------------------------
__global__ void k_init(float* m, float* denom, float* xagg, int nseg, int nagg) {
  int i = blockIdx.x * 256 + threadIdx.x;
  if (i < nseg) { m[i] = -3.0e38f; denom[i] = 0.f; }
  if (i < nagg) xagg[i] = 0.f;
}

// ---------------------------------------------------------------------------
// Naive small f32 GEMM: out[M,N] = (A[M,K] @ W) / rowdenom + bias
// transW==0: W is [K,N] row-major. transW==1: W is [N,K] row-major (use W^T).
// ---------------------------------------------------------------------------
__global__ void k_sgemm(const float* __restrict__ A, const float* __restrict__ W,
                        const float* __restrict__ bias, const float* __restrict__ rowdenom,
                        float* __restrict__ out, int N, int K, int transW) {
  __shared__ float sA[1024];
  const int row = blockIdx.x;
  for (int k = threadIdx.x; k < K; k += 256) sA[k] = A[(long)row * K + k];
  __syncthreads();
  float inv = 1.f;
  if (rowdenom) inv = 1.f / fmaxf(rowdenom[row], 1e-30f);
  for (int c = threadIdx.x; c < N; c += 256) {
    float acc = 0.f;
    if (transW) {
      const float* wr = W + (long)c * K;
      for (int k = 0; k < K; ++k) acc += sA[k] * wr[k];
    } else {
      for (int k = 0; k < K; ++k) acc += sA[k] * W[(long)k * N + c];
    }
    acc *= inv;
    if (bias) acc += bias[c];
    out[(long)row * N + c] = acc;
  }
}

// ---------------------------------------------------------------------------
// Scores: one wave per patch.
// s[b,p] = (QW[rid] . x[b,p] + qoff[rid]) * scale ; atomic segment max.
// ---------------------------------------------------------------------------
__global__ void k_scores(const float* __restrict__ x, const int* __restrict__ rid,
                         const float* __restrict__ QW, const float* __restrict__ qoff,
                         float* __restrict__ s, float* __restrict__ m) {
  const int wave = threadIdx.x >> 5, lane = threadIdx.x & 31;
  const long bp = (long)blockIdx.x * 8 + wave;        // grid = 4096 blocks
  const int r = rid[bp];
  const int b = (int)(bp >> 12);                      // /4096
  const float4* x4 = (const float4*)(x + bp * CC);
  const float4* q4 = (const float4*)(QW + (long)r * CC);
  float acc = 0.f;
#pragma unroll
  for (int i = 0; i < 6; ++i) {                       // 768/4 = 192 = 6*32
    float4 xa = x4[i * 32 + lane];
    float4 qa = q4[i * 32 + lane];
    acc += xa.x * qa.x + xa.y * qa.y + xa.z * qa.z + xa.w * qa.w;
  }
#pragma unroll
  for (int off = 16; off; off >>= 1) acc += __shfl_xor(acc, off, 32);
  if (lane == 0) {
    const float scale = 0.036084391824351614f;        // 1/sqrt(768)
    float sv = (acc + qoff[r]) * scale;
    s[bp] = sv;
    atomicMaxF(&m[b * NN + r], sv);
  }
}

// ---------------------------------------------------------------------------
// e = exp(s - m[seg]); denom[seg] += e; xagg[seg,:] += e * x[b,p,:]
// ---------------------------------------------------------------------------
__global__ void k_expacc(const float* __restrict__ x, const int* __restrict__ rid,
                         const float* __restrict__ s, const float* __restrict__ m,
                         float* __restrict__ denom, float* __restrict__ xagg) {
  const long bp = blockIdx.x;
  const int b = (int)(bp >> 12);
  const int r = rid[bp];
  const int seg = b * NN + r;
  const float e = __expf(s[bp] - m[seg]);
  if (threadIdx.x == 0) atomAddF(&denom[seg], e);
  const float* xr = x + bp * CC;
  float* agg = xagg + (long)seg * CC;
#pragma unroll
  for (int i = 0; i < 3; ++i) {
    int c = threadIdx.x + i * 256;
    atomAddF(&agg[c], e * xr[c]);
  }
}

// ---------------------------------------------------------------------------
// Yb[b,p,:] = bf16( x[b,p,:] + V2[b, rid[b,p], :] )
// ---------------------------------------------------------------------------
__global__ void k_gather_bf16(const float* __restrict__ x, const int* __restrict__ rid,
                              const float* __restrict__ V2, unsigned short* __restrict__ Yb) {
  const long bp = blockIdx.x;
  const int b = (int)(bp >> 12);
  const int r = rid[bp];
  const float* xr = x + bp * CC;
  const float* vr = V2 + (long)(b * NN + r) * CC;
  unsigned short* y = Yb + bp * CC;
#pragma unroll
  for (int i = 0; i < 3; ++i) {
    int c = threadIdx.x + i * 256;
    y[c] = f2bf(xr[c] + vr[c]);
  }
}

// ---------------------------------------------------------------------------
// Wo f32 [K,N] -> bf16 transposed Wt [N,K] (one-time, 1.2 MB)
// ---------------------------------------------------------------------------
__global__ void k_f2bf_t(const float* __restrict__ in, unsigned short* __restrict__ out) {
  int i = blockIdx.x * 256 + threadIdx.x;   // coalesced read of wo
  if (i < CC * CC) {
    int k = i / CC, n = i % CC;
    out[(long)n * CC + k] = f2bf(in[i]);
  }
}

// ---------------------------------------------------------------------------
// Main WMMA GEMM: out[M,N] = Yb[M,K](bf16) @ Wt[N,K](bf16, pre-transposed) + bias.
// M=32768, N=768, K=768. Block tile 128x128x32, 8 waves, wave tile 64x32
// (4x2 wmma 16x16). Double-buffered LDS, tiles staged with
// GLOBAL_LOAD_ASYNC_TO_LDS_B128 (ASYNCcnt) overlapping the WMMAs.
// ---------------------------------------------------------------------------
#define LDT 40   // LDS row stride in bf16 elems (80B, multiple of 16B)

__global__ void __launch_bounds__(256)
k_gemm_bf16(const unsigned short* __restrict__ Yb, const unsigned short* __restrict__ Wt,
            const float* __restrict__ bias, float* __restrict__ out) {
  __shared__ unsigned short lA[2][128 * LDT];
  __shared__ unsigned short lB[2][128 * LDT];

  const int bx = blockIdx.x;
  const int mTile = bx & 255;            // 256 M tiles
  const int nTile = bx >> 8;             // 6 N tiles
  const long mBase = (long)mTile * 128;
  const int nBase = nTile * 128;

  const int tid = threadIdx.x;
  const int lane = tid & 31, wave = tid >> 5;
  const int waveM = wave & 1, waveN = wave >> 1;    // 2 x 4 wave grid
  const int half = lane >> 4, l16 = lane & 15;

  // cooperative staging: both tiles are 128 rows x 32 bf16; thread -> 32B
  const int srow = tid >> 1, schunk = (tid & 1) * 16;
  const unsigned short* gA = Yb + (mBase + srow) * CC + schunk;
  const unsigned short* gB = Wt + (long)(nBase + srow) * CC + schunk;

  auto stage = [&](int buf, int kb) {
    unsigned aoff = (unsigned)(uintptr_t)&lA[buf][srow * LDT + schunk];
    unsigned boff = (unsigned)(uintptr_t)&lB[buf][srow * LDT + schunk];
    asyncB128(aoff, gA + kb);
    asyncB128_o16(aoff, gA + kb);       // INST_OFFSET applies to both sides
    asyncB128(boff, gB + kb);
    asyncB128_o16(boff, gB + kb);
  };

  v8f acc[4][2];
#pragma unroll
  for (int mi = 0; mi < 4; ++mi)
#pragma unroll
    for (int ni = 0; ni < 2; ++ni) acc[mi][ni] = (v8f){0.f,0.f,0.f,0.f,0.f,0.f,0.f,0.f};

  union AU { v16bf v; unsigned u[8]; };
  union BU { v16bf v; uint4 q[2]; };

  stage(0, 0);
  waitAsync0();
  __syncthreads();

  int buf = 0;
  for (int kb = 0; kb < CC; kb += 32) {
    if (kb + 32 < CC) stage(buf ^ 1, kb + 32);   // DMA next tile during WMMAs

    // ---- fragments from current buffer ----
    v16bf af[4], bfr[2];
#pragma unroll
    for (int mi = 0; mi < 4; ++mi) {
      const int row = waveM * 64 + mi * 16 + l16;
      AU a;
#pragma unroll
      for (int v = 0; v < 8; ++v) {
        const int ke = ((v < 4) ? 2 * v : 2 * v + 8) + 8 * half;  // ISA A-layout interleave
        a.u[v] = *(const unsigned*)&lA[buf][row * LDT + ke];
      }
      af[mi] = a.v;
    }
#pragma unroll
    for (int ni = 0; ni < 2; ++ni) {
      const int nl = waveN * 32 + ni * 16 + l16;
      BU b;
      b.q[0] = *(const uint4*)&lB[buf][nl * LDT + half * 8];
      b.q[1] = *(const uint4*)&lB[buf][nl * LDT + half * 8 + 16];
      bfr[ni] = b.v;
    }

    // ---- 8 WMMAs ----
#pragma unroll
    for (int mi = 0; mi < 4; ++mi)
#pragma unroll
      for (int ni = 0; ni < 2; ++ni)
        acc[mi][ni] = __builtin_amdgcn_wmma_f32_16x16x32_bf16(
            false, af[mi], false, bfr[ni], (short)0, acc[mi][ni], false, false);

    waitAsync0();       // own async copies of next tile done
    __syncthreads();    // everyone's copies + LDS reads done
    buf ^= 1;
  }

  // ---- epilogue: bias + store ----
#pragma unroll
  for (int ni = 0; ni < 2; ++ni) {
    const int col = nBase + waveN * 32 + ni * 16 + l16;
    const float bb = bias[col];
#pragma unroll
    for (int mi = 0; mi < 4; ++mi) {
#pragma unroll
      for (int r = 0; r < 8; ++r) {
        const long row = mBase + waveM * 64 + mi * 16 + r + 8 * half;
        out[row * CC + col] = acc[mi][ni][r] + bb;
      }
    }
  }
}

// ---------------------------------------------------------------------------
extern "C" void kernel_launch(void* const* d_in, const int* in_sizes, int n_in,
                              void* d_out, int out_size, void* d_ws, size_t ws_size,
                              hipStream_t stream) {
  const float* x      = (const float*)d_in[0];
  const int*   rid    = (const int*)  d_in[1];
  const float* rtok   = (const float*)d_in[2];
  const float* wq_r   = (const float*)d_in[3];
  const float* bq_r   = (const float*)d_in[4];
  const float* wk_p   = (const float*)d_in[5];
  const float* bk_p   = (const float*)d_in[6];
  const float* wv_p   = (const float*)d_in[7];
  const float* bv_p   = (const float*)d_in[8];
  // d_in[9..12]: wq_p,bq_p,wk_r,bk_r are mathematically dead (see reference)
  const float* wv_r   = (const float*)d_in[13];
  const float* bv_r   = (const float*)d_in[14];
  const float* wo     = (const float*)d_in[15];
  const float* bo     = (const float*)d_in[16];
  float* out = (float*)d_out;

  // workspace carve-up (bytes, 256-aligned)
  char* base = (char*)d_ws;
  size_t off = 0;
  auto carve = [&](size_t bytes) { char* p = base + off; off = (off + bytes + 255) & ~(size_t)255; return p; };
  float*          Q     = (float*)         carve((size_t)NN * CC * 4);
  float*          QW    = (float*)         carve((size_t)NN * CC * 4);
  float*          qoff  = (float*)         carve((size_t)NN * 4);
  float*          s     = (float*)         carve((size_t)BP * 4);
  float*          m     = (float*)         carve((size_t)SEG * 4);
  float*          denom = (float*)         carve((size_t)SEG * 4);
  float*          xagg  = (float*)         carve((size_t)SEG * CC * 4);
  float*          rg    = (float*)         carve((size_t)SEG * CC * 4);
  float*          V2    = (float*)         carve((size_t)SEG * CC * 4);
  unsigned short* Yb    = (unsigned short*)carve((size_t)BP * CC * 2);
  unsigned short* Wt    = (unsigned short*)carve((size_t)CC * CC * 2);
  (void)ws_size; (void)n_in; (void)in_sizes; (void)out_size;

  // 1) Q = region_tokens @ wq_r + bq_r                        [116,768]
  k_sgemm<<<NN, 256, 0, stream>>>(rtok, wq_r, bq_r, nullptr, Q, CC, CC, 0);
  // 2) QW = Q @ wk_p^T                                        [116,768]
  k_sgemm<<<NN, 256, 0, stream>>>(Q, wk_p, nullptr, nullptr, QW, CC, CC, 1);
  // 3) qoff = Q @ bk_p                                        [116]
  k_sgemm<<<NN, 256, 0, stream>>>(Q, bk_p, nullptr, nullptr, qoff, 1, CC, 0);
  // 4) init segment accumulators
  {
    int nagg = SEG * CC;
    k_init<<<(nagg + 255) / 256, 256, 0, stream>>>(m, denom, xagg, SEG, nagg);
  }
  // 5) scores + segment max (one wave per patch)
  k_scores<<<BP / 8, 256, 0, stream>>>(x, rid, QW, qoff, s, m);
  // 6) exp + segment denom + weighted x aggregation
  k_expacc<<<BP, 256, 0, stream>>>(x, rid, s, m, denom, xagg);
  // 7) region_out = (xagg/denom) @ wv_p + bv_p                [928,768]
  k_sgemm<<<SEG, 256, 0, stream>>>(xagg, wv_p, bv_p, denom, rg, CC, CC, 0);
  // 8) V2 = region_out @ wv_r + bv_r                          [928,768]
  k_sgemm<<<SEG, 256, 0, stream>>>(rg, wv_r, bv_r, nullptr, V2, CC, CC, 0);
  // 9) Yb = bf16(x + gather(V2)); Wt = bf16(wo)^T
  k_gather_bf16<<<BP, 256, 0, stream>>>(x, rid, V2, Yb);
  k_f2bf_t<<<(CC * CC + 255) / 256, 256, 0, stream>>>(wo, Wt);
  // 10) out = Yb @ Wt^T + bo  (the one big GEMM, bf16 WMMA + async LDS DMA)
  k_gemm_bf16<<<(BP / 128) * (CC / 128), 256, 0, stream>>>(Yb, Wt, bo, out);
}
